// MessagePassingLayer_28295244546255
// MI455X (gfx1250) — matrix operand order repaired and checked by reading, not compiled, not run on hardware
//
#include <hip/hip_runtime.h>
#include <hip/hip_bf16.h>
#include <stddef.h>

#define ND 128
#define ED 64
#define KIN 320          // 2*ND + ED
#define WPB 4            // waves per block (wave32)
#define LN_EPS 1e-5f

typedef __attribute__((ext_vector_type(16))) __bf16 v16bf;
typedef __attribute__((ext_vector_type(8)))  float  v8f;
typedef __attribute__((ext_vector_type(4)))  float  f32x4;

__device__ __forceinline__ v8f wmma_bf16(v16bf a, v16bf b, v8f c) {
  // D = A(16x32 bf16) x B(32x16 bf16) + C(16x16 f32)
  return __builtin_amdgcn_wmma_f32_16x16x32_bf16(
      /*neg_a=*/false, a, /*neg_b=*/false, b,
      /*c_mod=*/(short)0, c, /*reuse_a=*/false, /*reuse_b=*/false);
}

// A fragment from LDS, tile row-major [16][Ktot] bf16.
// Contiguous k order: lanes 0-15 hold row m, K 0..15;
// lanes 16-31 hold row m, K 16..31.
__device__ __forceinline__ v16bf load_fragA(const __bf16* base, int Ktot, int kt, int lane) {
  const int m  = lane & 15;
  const int hk = lane >> 4;
  const __bf16* row = base + m * Ktot + 32 * kt + 16 * hk;
  union { v16bf v; f32x4 f[2]; } u;
  u.f[0] = *(const f32x4*)(row);
  u.f[1] = *(const f32x4*)(row + 8);
  return u.v;
}

// B fragment from global, weights stored transposed row-major [N][Ktot] bf16.
// Contiguous k order: lanes 0-15 hold col n, K 0..15; lanes 16-31 hold col n, K 16..31.
__device__ __forceinline__ v16bf load_fragB(const __bf16* bt, int Ktot, int nbase, int kt, int lane) {
  const int n  = nbase + (lane & 15);
  const int hk = lane >> 4;
  const __bf16* p = bt + (size_t)n * Ktot + 32 * kt + 16 * hk;
  union { v16bf v; f32x4 f[2]; } u;
  u.f[0] = ((const f32x4*)p)[0];
  u.f[1] = ((const f32x4*)p)[1];
  return u.v;
}

// Convert float4 -> 4x bf16, packed 8-byte LDS store.
__device__ __forceinline__ void store_bf4(__bf16* p, f32x4 v) {
  union { __bf16 h[4]; unsigned long long u; } x;
  x.h[0] = (__bf16)v.x; x.h[1] = (__bf16)v.y;
  x.h[2] = (__bf16)v.z; x.h[3] = (__bf16)v.w;
  *(unsigned long long*)p = x.u;
}

__device__ __forceinline__ float sigmoidf_(float x) { return 1.0f / (1.0f + __expf(-x)); }

__global__ void zero_f32_kernel(float* __restrict__ p, int n) {
  int i = blockIdx.x * blockDim.x + threadIdx.x;
  if (i < n) p[i] = 0.0f;
}

// Build bf16 weight layouts in workspace:
//   w1t  [128][320] = W1^T,  w2t [128][128] = W2^T  (B = W, B^T row-major)
//   wihb [384][128] = W_ih,  whhb[384][128] = W_hh  (B = W^T, B^T = W row-major)
__global__ void convert_weights_kernel(
    const float* __restrict__ W1, const float* __restrict__ W2,
    const float* __restrict__ Wih, const float* __restrict__ Whh,
    __bf16* __restrict__ w1t, __bf16* __restrict__ w2t,
    __bf16* __restrict__ wihb, __bf16* __restrict__ whhb) {
  int i = blockIdx.x * blockDim.x + threadIdx.x;
  if (i < KIN * ND) { int n = i / KIN, k = i % KIN; w1t[i] = (__bf16)W1[k * ND + n]; }
  if (i < ND * ND)  { int n = i / ND,  k = i % ND;  w2t[i] = (__bf16)W2[k * ND + n]; }
  if (i < 3 * ND * ND) { wihb[i] = (__bf16)Wih[i]; whhb[i] = (__bf16)Whh[i]; }
}

// One wave = one 16-edge tile: gather -> GEMM1(WMMA) -> in-register LayerNorm/ReLU
// -> GEMM2(WMMA) -> gate -> f32 atomic scatter-add into agg.
__global__ __launch_bounds__(WPB * 32) void edge_message_kernel(
    const float* __restrict__ nodef,
    const int*   __restrict__ eidx,
    const float* __restrict__ edgef,
    const __bf16* __restrict__ w1t,
    const float* __restrict__ b1,
    const float* __restrict__ ln_g,
    const float* __restrict__ ln_b,
    const __bf16* __restrict__ w2t,
    const float* __restrict__ b2,
    const float* __restrict__ gate_w,
    const float* __restrict__ gate_b,
    float* __restrict__ agg,
    int E) {
  __shared__ __align__(16) __bf16 Abuf[WPB][16 * KIN];   // 40 KB
  __shared__ __align__(16) __bf16 A2buf[WPB][16 * ND];   // 16 KB
  __shared__ int   srcbuf[WPB][16];
  __shared__ int   dstbuf[WPB][16];
  __shared__ float gatebuf[WPB][16];

  const int lane = threadIdx.x & 31;
  const int w    = threadIdx.x >> 5;
  const int hk   = lane >> 4;
  const int cb   = lane & 15;
  const int e0   = (blockIdx.x * WPB + w) * 16;

  const int* __restrict__ srcp = eidx;
  const int* __restrict__ dstp = eidx + E;

  if (threadIdx.x == 0) { __builtin_prefetch(w1t, 0, 0); __builtin_prefetch(w2t, 0, 0); }

  // edge ids + sigmoid gate (one edge per lane 0..15), vectorized dot
  if (lane < 16) {
    int e = e0 + lane;
    srcbuf[w][lane] = srcp[e];
    dstbuf[w][lane] = dstp[e];
    const f32x4* ef4 = (const f32x4*)(edgef + (size_t)e * ED);
    const f32x4* gw4 = (const f32x4*)gate_w;
    float acc = gate_b[0];
#pragma unroll
    for (int j = 0; j < ED / 4; ++j) {
      f32x4 a4 = ef4[j], b4 = gw4[j];
      acc += a4.x * b4.x + a4.y * b4.y + a4.z * b4.z + a4.w * b4.w;
    }
    gatebuf[w][lane] = sigmoidf_(acc);
  }
  __syncthreads();

  // gather [src | dst | edge], f32 -> bf16, vectorized b128 loads / b64 LDS stores
  for (int m = 0; m < 16; ++m) {
    const f32x4* sp4 = (const f32x4*)(nodef + (size_t)srcbuf[w][m] * ND);
    const f32x4* dp4 = (const f32x4*)(nodef + (size_t)dstbuf[w][m] * ND);
    store_bf4(&Abuf[w][m * KIN + 4 * lane], sp4[lane]);
    store_bf4(&Abuf[w][m * KIN + ND + 4 * lane], dp4[lane]);
    if (lane < 16) {
      const f32x4* ep4 = (const f32x4*)(edgef + (size_t)(e0 + m) * ED);
      store_bf4(&Abuf[w][m * KIN + 2 * ND + 4 * lane], ep4[lane]);
    }
  }
  __syncthreads();

  // GEMM1: [16,320] @ [320,128], f32 accum seeded with b1.
  // B fragments double-buffered so wmma(t) overlaps load of B(t+1).
  v8f acc[8];
#pragma unroll
  for (int t = 0; t < 8; ++t) {
    float bv = b1[16 * t + cb];
#pragma unroll
    for (int i = 0; i < 8; ++i) acc[t][i] = bv;
  }
  for (int kt = 0; kt < KIN / 32; ++kt) {
    v16bf a = load_fragA(&Abuf[w][0], KIN, kt, lane);
    v16bf b = load_fragB(w1t, KIN, 0, kt, lane);
#pragma unroll
    for (int t = 0; t < 8; ++t) {
      v16bf bn = b;
      if (t < 7) bn = load_fragB(w1t, KIN, 16 * (t + 1), kt, lane);
      acc[t] = wmma_bf16(a, b, acc[t]);
      b = bn;
    }
  }

  // In-register LayerNorm: each row m = r + 8*hk lives in VGPR r of one
  // 16-lane half across 8 n-tiles -> half-wave shfl_xor reduction.
  float s1[8], s2[8];
#pragma unroll
  for (int r = 0; r < 8; ++r) { s1[r] = 0.f; s2[r] = 0.f; }
#pragma unroll
  for (int t = 0; t < 8; ++t) {
#pragma unroll
    for (int r = 0; r < 8; ++r) { float x = acc[t][r]; s1[r] += x; s2[r] += x * x; }
  }
#pragma unroll
  for (int r = 0; r < 8; ++r) {
#pragma unroll
    for (int mask = 1; mask < 16; mask <<= 1) {
      s1[r] += __shfl_xor(s1[r], mask, 32);
      s2[r] += __shfl_xor(s2[r], mask, 32);
    }
  }
  float mean[8], rstd[8];
#pragma unroll
  for (int r = 0; r < 8; ++r) {
    float m = s1[r] * (1.0f / ND);
    float v = s2[r] * (1.0f / ND) - m * m;
    mean[r] = m;
    rstd[r] = rsqrtf(v + LN_EPS);
  }
#pragma unroll
  for (int t = 0; t < 8; ++t) {
    float g  = ln_g[16 * t + cb];
    float bb = ln_b[16 * t + cb];
#pragma unroll
    for (int r = 0; r < 8; ++r) {
      float x = (acc[t][r] - mean[r]) * rstd[r] * g + bb;
      x = fmaxf(x, 0.0f);                                   // ReLU
      A2buf[w][(r + 8 * hk) * ND + 16 * t + cb] = (__bf16)x;
    }
  }
  __syncthreads();

  // GEMM2: [16,128] @ [128,128] + b2, then gate * message -> atomic scatter-add
  v16bf a2[4];
#pragma unroll
  for (int kt = 0; kt < 4; ++kt) a2[kt] = load_fragA(&A2buf[w][0], ND, kt, lane);
  float gv[8];
  int   dv[8];
#pragma unroll
  for (int r = 0; r < 8; ++r) {
    int m = r + 8 * hk;
    gv[r] = gatebuf[w][m];
    dv[r] = dstbuf[w][m];
  }
#pragma unroll
  for (int nt = 0; nt < 8; ++nt) {
    v8f c;
    float bv = b2[16 * nt + cb];
#pragma unroll
    for (int i = 0; i < 8; ++i) c[i] = bv;
    v16bf b = load_fragB(w2t, ND, 16 * nt, 0, lane);
#pragma unroll
    for (int kt = 0; kt < 4; ++kt) {
      v16bf bn = b;
      if (kt < 3) bn = load_fragB(w2t, ND, 16 * nt, kt + 1, lane);
      c = wmma_bf16(a2[kt], b, c);
      b = bn;
    }
#pragma unroll
    for (int r = 0; r < 8; ++r) {
      atomicAdd(&agg[(size_t)dv[r] * ND + 16 * nt + cb], c[r] * gv[r]);
    }
  }
}

// One wave = one 16-node tile: gi = agg @ Wih^T, gh = h @ Whh^T (6 WMMA tiles
// per 16-col slab), fused GRU gate epilogue (torch order r,z,n).
__global__ __launch_bounds__(WPB * 32) void gru_update_kernel(
    const float* __restrict__ agg,
    const float* __restrict__ nodef,
    const __bf16* __restrict__ wih,
    const float* __restrict__ bih,
    const __bf16* __restrict__ whh,
    const float* __restrict__ bhh,
    float* __restrict__ out) {
  __shared__ __align__(16) __bf16 Ai[WPB][16 * ND];
  __shared__ __align__(16) __bf16 Ah[WPB][16 * ND];
  const int lane = threadIdx.x & 31;
  const int w    = threadIdx.x >> 5;
  const int hk   = lane >> 4;
  const int cb   = lane & 15;
  const int n0   = (blockIdx.x * WPB + w) * 16;

  for (int m = 0; m < 16; ++m) {
    const f32x4* ap4 = (const f32x4*)(agg   + (size_t)(n0 + m) * ND);
    const f32x4* hp4 = (const f32x4*)(nodef + (size_t)(n0 + m) * ND);
    store_bf4(&Ai[w][m * ND + 4 * lane], ap4[lane]);
    store_bf4(&Ah[w][m * ND + 4 * lane], hp4[lane]);
  }
  __syncthreads();

  v16bf ai[4], ah[4];
#pragma unroll
  for (int kt = 0; kt < 4; ++kt) {
    ai[kt] = load_fragA(&Ai[w][0], ND, kt, lane);
    ah[kt] = load_fragA(&Ah[w][0], ND, kt, lane);
  }

#pragma unroll
  for (int nt = 0; nt < 8; ++nt) {
    v8f cir, ciz, cin2, chr2, chz2, chn2;
#pragma unroll
    for (int i = 0; i < 8; ++i) {
      cir[i] = 0.f; ciz[i] = 0.f; cin2[i] = 0.f;
      chr2[i] = 0.f; chz2[i] = 0.f; chn2[i] = 0.f;
    }
#pragma unroll
    for (int kt = 0; kt < 4; ++kt) {
      cir  = wmma_bf16(ai[kt], load_fragB(wih, ND, 0      + 16 * nt, kt, lane), cir);
      ciz  = wmma_bf16(ai[kt], load_fragB(wih, ND, ND     + 16 * nt, kt, lane), ciz);
      cin2 = wmma_bf16(ai[kt], load_fragB(wih, ND, 2 * ND + 16 * nt, kt, lane), cin2);
      chr2 = wmma_bf16(ah[kt], load_fragB(whh, ND, 0      + 16 * nt, kt, lane), chr2);
      chz2 = wmma_bf16(ah[kt], load_fragB(whh, ND, ND     + 16 * nt, kt, lane), chz2);
      chn2 = wmma_bf16(ah[kt], load_fragB(whh, ND, 2 * ND + 16 * nt, kt, lane), chn2);
    }
    const int c = 16 * nt + cb;
    float bir = bih[c], biz = bih[ND + c], bin = bih[2 * ND + c];
    float bhr = bhh[c], bhz = bhh[ND + c], bhn = bhh[2 * ND + c];
#pragma unroll
    for (int r = 0; r < 8; ++r) {
      int node = n0 + r + 8 * hk;
      float rg = sigmoidf_((cir[r] + bir) + (chr2[r] + bhr));
      float zg = sigmoidf_((ciz[r] + biz) + (chz2[r] + bhz));
      float ng = tanhf((cin2[r] + bin) + rg * (chn2[r] + bhn));
      float hv = nodef[(size_t)node * ND + c];
      out[(size_t)node * ND + c] = (1.0f - zg) * ng + zg * hv;
    }
  }
}

extern "C" void kernel_launch(void* const* d_in, const int* in_sizes, int n_in,
                              void* d_out, int out_size, void* d_ws, size_t ws_size,
                              hipStream_t stream) {
  (void)n_in; (void)out_size; (void)ws_size;
  const float* nodef  = (const float*)d_in[0];
  const int*   eidx   = (const int*)d_in[1];
  const float* edgef  = (const float*)d_in[2];
  const float* W1     = (const float*)d_in[3];
  const float* b1     = (const float*)d_in[4];
  const float* ln_g   = (const float*)d_in[5];
  const float* ln_b   = (const float*)d_in[6];
  const float* W2     = (const float*)d_in[7];
  const float* b2     = (const float*)d_in[8];
  const float* gate_w = (const float*)d_in[9];
  const float* gate_b = (const float*)d_in[10];
  const float* Wih    = (const float*)d_in[11];
  const float* bih    = (const float*)d_in[12];
  const float* Whh    = (const float*)d_in[13];
  const float* bhh    = (const float*)d_in[14];
  float* out = (float*)d_out;

  const int Nn = in_sizes[0] / ND;   // 40000
  const int E  = in_sizes[1] / 2;    // 640000

  // workspace: [agg f32 N*ND][w1t][w2t][wih][whh] (bf16)
  char* ws = (char*)d_ws;
  float* agg = (float*)ws;
  size_t aggBytes = (size_t)Nn * ND * sizeof(float);   // 16B aligned for these sizes
  __bf16* w1t  = (__bf16*)(ws + aggBytes);
  __bf16* w2t  = w1t  + KIN * ND;
  __bf16* wihb = w2t  + ND * ND;
  __bf16* whhb = wihb + 3 * ND * ND;

  zero_f32_kernel<<<(Nn * ND + 255) / 256, 256, 0, stream>>>(agg, Nn * ND);
  convert_weights_kernel<<<(3 * ND * ND + 255) / 256, 256, 0, stream>>>(
      W1, W2, Wih, Whh, w1t, w2t, wihb, whhb);
  edge_message_kernel<<<E / (16 * WPB), WPB * 32, 0, stream>>>(
      nodef, eidx, edgef, w1t, b1, ln_g, ln_b, w2t, b2, gate_w, gate_b, agg, E);
  gru_update_kernel<<<Nn / (16 * WPB), WPB * 32, 0, stream>>>(
      agg, nodef, wihb, bih, whhb, bhh, out);
}
